// TrackFormerModel_84705345012198
// MI455X (gfx1250) — matrix operand order repaired and checked by reading, not compiled, not run on hardware
//
#include <hip/hip_runtime.h>
#include <hip/hip_bf16.h>

typedef __attribute__((ext_vector_type(16))) __bf16 v16bf;
typedef __attribute__((ext_vector_type(8)))  float  v8f;

#define N_ 4096
#define M_ 10240
#define D_ 256
#define ALPHA_ 0.7f
#define BETA_  0.3f
#define KSTEPS     (D_ / 32)          // 8 WMMA K-steps of 32
#define ROW_TILES  (N_ / 16)          // 256
#define COL_TILES  (M_ / 16)          // 640
#define COLS_PER_WAVE 4               // each wave -> 16x64 output tile
#define COL_GROUPS (COL_TILES / COLS_PER_WAVE)  // 160
#define NWAVES     (ROW_TILES * COL_GROUPS)     // 40960 full waves

// ---------------------------------------------------------------------------
// Stage curr_embeds (fp32, row-major NxD) into WMMA A-operand layout, split
// into bf16 hi + lo. ISA 16-bit A 16x32 layout (wave32):
//   lanes 0-15 : M=lane,    K order [0..7, 16..23]
//   lanes 16-31: M=lane-16, K order [8..15, 24..31]
// Flat stage index o = (((it*8 + kk)*32 + lane)*16 + idx)
// ---------------------------------------------------------------------------
__global__ void stage_a_kernel(const float* __restrict__ curr,
                               __bf16* __restrict__ a_hi,
                               __bf16* __restrict__ a_lo) {
  int o    = blockIdx.x * blockDim.x + threadIdx.x;   // 0 .. N_*D_-1
  int idx  = o & 15;
  int lane = (o >> 4) & 31;
  int kk   = (o >> 9) & 7;
  int it   = o >> 12;
  int m = it * 16 + (lane & 15);
  int k;
  if (lane < 16) k = kk * 32 + (idx < 8 ? idx      : idx + 8);
  else           k = kk * 32 + (idx < 8 ? idx + 8  : idx + 16);
  float x = curr[(size_t)m * D_ + k];
  __bf16 h = (__bf16)x;
  a_hi[o] = h;
  a_lo[o] = (__bf16)(x - (float)h);
}

// ---------------------------------------------------------------------------
// Stage prev_embeds (fp32, row-major MxD) into WMMA B-operand layout
// (transpose happens here). ISA 16-bit B 32x16 layout (wave32):
//   lanes 0-15 : N=lane,    K order [0..7, 16..23]
//   lanes 16-31: N=lane-16, K order [8..15, 24..31]
// ---------------------------------------------------------------------------
__global__ void stage_b_kernel(const float* __restrict__ prev,
                               __bf16* __restrict__ b_hi,
                               __bf16* __restrict__ b_lo) {
  int o    = blockIdx.x * blockDim.x + threadIdx.x;   // 0 .. M_*D_-1
  int idx  = o & 15;
  int lane = (o >> 4) & 31;
  int kk   = (o >> 9) & 7;
  int jt   = o >> 12;
  int n = jt * 16 + (lane & 15);
  int k = kk * 32 + (idx < 8 ? idx : idx + 8) + (lane < 16 ? 0 : 8);
  float x = prev[(size_t)n * D_ + k];
  __bf16 h = (__bf16)x;
  b_hi[o] = h;
  b_lo[o] = (__bf16)(x - (float)h);
}

// ---------------------------------------------------------------------------
// Per-row precompute: exact fp32 ||x||^2, clamped norm, box centers.
// ---------------------------------------------------------------------------
__global__ void stage_rows_kernel(const float* __restrict__ emb,
                                  const float* __restrict__ boxes, int rows,
                                  float* __restrict__ nrm, float* __restrict__ n2,
                                  float* __restrict__ cx,  float* __restrict__ cy) {
  int r = blockIdx.x * blockDim.x + threadIdx.x;
  if (r >= rows) return;
  const float4* e = (const float4*)(emb + (size_t)r * D_);
  float s = 0.f;
  #pragma unroll 4
  for (int q = 0; q < D_ / 4; ++q) {
    float4 v = e[q];
    s += v.x * v.x + v.y * v.y + v.z * v.z + v.w * v.w;
  }
  n2[r]  = s;
  nrm[r] = fmaxf(sqrtf(s), 1e-12f);
  float bx = boxes[r * 4 + 0], by = boxes[r * 4 + 1];
  float bw = boxes[r * 4 + 2], bh = boxes[r * 4 + 3];
  cx[r] = bx + 0.5f * bw;
  cy[r] = by + 0.5f * bh;
}

// ---------------------------------------------------------------------------
// Fused WMMA GEMM + epilogue. One wave computes a 16x64 tile of G=curr@prev^T
// with bf16x3 emulation (hi*hi + hi*lo + lo*hi), then writes total_cost and
// accumulates the contrastive-loss partial for its tile.
// ---------------------------------------------------------------------------
__global__ __launch_bounds__(256) void gemm_epilogue_kernel(
    const __bf16* __restrict__ a_hi, const __bf16* __restrict__ a_lo,
    const __bf16* __restrict__ b_hi, const __bf16* __restrict__ b_lo,
    const float* __restrict__ ncn, const float* __restrict__ nc2,
    const float* __restrict__ ccx, const float* __restrict__ ccy,
    const float* __restrict__ npn, const float* __restrict__ np2,
    const float* __restrict__ pcx, const float* __restrict__ pcy,
    const int* __restrict__ cid, const int* __restrict__ pid,
    float* __restrict__ out, float* __restrict__ partials) {
  int gid  = blockIdx.x * blockDim.x + threadIdx.x;
  int wave = gid >> 5;
  int lane = gid & 31;
  int it = wave / COL_GROUPS;         // row tile   0..255
  int jg = wave % COL_GROUPS;         // col group  0..159 (4 tiles each)

  const v16bf* Ah = (const v16bf*)a_hi;
  const v16bf* Al = (const v16bf*)a_lo;
  const v16bf* Bh = (const v16bf*)b_hi;
  const v16bf* Bl = (const v16bf*)b_lo;

  v8f acc[COLS_PER_WAVE];
  #pragma unroll
  for (int t = 0; t < COLS_PER_WAVE; ++t) acc[t] = (v8f){};

  for (int kk = 0; kk < KSTEPS; ++kk) {
    int abase = (it * KSTEPS + kk) * 32 + lane;
    v16bf ah = Ah[abase];
    v16bf al = Al[abase];
    #pragma unroll
    for (int t = 0; t < COLS_PER_WAVE; ++t) {
      int jt = jg * COLS_PER_WAVE + t;
      int bbase = (jt * KSTEPS + kk) * 32 + lane;
      v16bf bh = Bh[bbase];
      v16bf bl = Bl[bbase];
      acc[t] = __builtin_amdgcn_wmma_f32_16x16x32_bf16(
          false, ah, false, bh, (short)0, acc[t], false, false);
      acc[t] = __builtin_amdgcn_wmma_f32_16x16x32_bf16(
          false, ah, false, bl, (short)0, acc[t], false, false);
      acc[t] = __builtin_amdgcn_wmma_f32_16x16x32_bf16(
          false, al, false, bh, (short)0, acc[t], false, false);
    }
  }

  // C/D layout: lane l, VGPR r  ->  M = r + 8*(l>=16), N = l&15
  int col_in = lane & 15;
  int rbase  = it * 16 + ((lane >> 4) << 3);
  float lsum = 0.f;
  #pragma unroll
  for (int t = 0; t < COLS_PER_WAVE; ++t) {
    int j = (jg * COLS_PER_WAVE + t) * 16 + col_in;
    float npn_j = npn[j], np2_j = np2[j];
    float px = pcx[j], py = pcy[j];
    int id_j = pid[j];
    #pragma unroll
    for (int r = 0; r < 8; ++r) {
      int i = rbase + r;
      float dot = acc[t][r];
      // total_cost
      float cosv  = dot / (ncn[i] * npn_j);
      float dx = ccx[i] - px, dy = ccy[i] - py;
      float cdist = sqrtf(dx * dx + dy * dy);
      out[(size_t)i * M_ + j] = ALPHA_ * (1.f - cosv) + BETA_ * cdist;
      // loss contribution
      float d2   = fmaxf(nc2[i] + np2_j - 2.f * dot, 0.f);
      float dist = sqrtf(d2);
      float hng  = fmaxf(1.f - dist, 0.f);
      lsum += (cid[i] == id_j) ? d2 : hng * hng;
    }
  }
  // wave32 reduce, deterministic
  #pragma unroll
  for (int off = 16; off > 0; off >>= 1) lsum += __shfl_xor(lsum, off, 32);
  if (lane == 0) partials[wave] = lsum;
}

// Fixed-order final reduction -> deterministic loss.
__global__ void reduce_loss_kernel(const float* __restrict__ partials,
                                   float* __restrict__ out_loss) {
  __shared__ float s[256];
  float v = 0.f;
  for (int i = threadIdx.x; i < NWAVES; i += 256) v += partials[i];
  s[threadIdx.x] = v;
  __syncthreads();
  for (int st = 128; st > 0; st >>= 1) {
    if ((int)threadIdx.x < st) s[threadIdx.x] += s[threadIdx.x + st];
    __syncthreads();
  }
  if (threadIdx.x == 0) *out_loss = s[0] / ((float)N_ * (float)M_);
}

extern "C" void kernel_launch(void* const* d_in, const int* in_sizes, int n_in,
                              void* d_out, int out_size, void* d_ws, size_t ws_size,
                              hipStream_t stream) {
  (void)in_sizes; (void)n_in; (void)out_size; (void)ws_size;
  const float* curr_e = (const float*)d_in[0];
  const float* curr_b = (const float*)d_in[1];
  const float* prev_e = (const float*)d_in[2];
  const float* prev_b = (const float*)d_in[3];
  const int*   cid    = (const int*)d_in[4];
  const int*   pid    = (const int*)d_in[5];
  float* out = (float*)d_out;   // [N*M total_cost] ++ [1 loss]

  // Workspace layout (~14.4 MB total; fits easily in L2 during the GEMM).
  __bf16* a_hi = (__bf16*)d_ws;
  __bf16* a_lo = a_hi + (size_t)N_ * D_;
  __bf16* b_hi = a_lo + (size_t)N_ * D_;
  __bf16* b_lo = b_hi + (size_t)M_ * D_;
  float* f = (float*)(b_lo + (size_t)M_ * D_);
  float* ncn = f; f += N_;
  float* nc2 = f; f += N_;
  float* ccx = f; f += N_;
  float* ccy = f; f += N_;
  float* npn = f; f += M_;
  float* np2 = f; f += M_;
  float* pcx = f; f += M_;
  float* pcy = f; f += M_;
  float* partials = f;          // NWAVES floats

  stage_a_kernel<<<(N_ * D_) / 256, 256, 0, stream>>>(curr_e, a_hi, a_lo);
  stage_b_kernel<<<(M_ * D_) / 256, 256, 0, stream>>>(prev_e, b_hi, b_lo);
  stage_rows_kernel<<<(N_ + 255) / 256, 256, 0, stream>>>(curr_e, curr_b, N_,
                                                          ncn, nc2, ccx, ccy);
  stage_rows_kernel<<<(M_ + 255) / 256, 256, 0, stream>>>(prev_e, prev_b, M_,
                                                          npn, np2, pcx, pcy);
  gemm_epilogue_kernel<<<(NWAVES * 32) / 256, 256, 0, stream>>>(
      a_hi, a_lo, b_hi, b_lo,
      ncn, nc2, ccx, ccy, npn, np2, pcx, pcy,
      cid, pid, out, partials);
  reduce_loss_kernel<<<1, 256, 0, stream>>>(partials, out + (size_t)N_ * M_);
}
